// ContextualAttentionLayer_69131793596956
// MI455X (gfx1250) — matrix operand order repaired and checked by reading, not compiled, not run on hardware
//
#include <hip/hip_runtime.h>
#include <hip/hip_bf16.h>

typedef __attribute__((ext_vector_type(16))) __bf16 v16bf;
typedef __attribute__((ext_vector_type(8)))  float  v8f;

#define ATT    256
#define GENES  2128
#define HID    512
#define BB     128
#define TT     1024

// ---------------------------------------------------------------------------
// Kernel 1: pack dense_kernel [512,256] f32 -> bf16 in WMMA B-operand layout.
// B tile for v_wmma_f32_16x16x32_bf16 is 32(K) x 16(N):
//   lane = ((klocal/8)%2)*16 + (n%16), element e = (klocal/16)*8 + klocal%8   (ISA 7.12.2)
// Tiles stored as [nt(16)][ks(16)] blocks of 32 lanes * 16 bf16 = 1KB each.
// ---------------------------------------------------------------------------
__global__ void pack_b_kernel(const float* __restrict__ dk, __bf16* __restrict__ packB) {
    int h = blockIdx.x;        // 0..511  (K dim of GEMM)
    int a = threadIdx.x;       // 0..255  (N dim of GEMM)
    int nt = a >> 4, nl = a & 15;
    int ks = h >> 5, kl = h & 31;
    int hl = (kl >> 3) & 1, e = ((kl >> 4) << 3) + (kl & 7);
    int lane = hl * 16 + nl;
    size_t idx = ((size_t)(nt * 16 + ks) * 32 + lane) * 16 + e;
    packB[idx] = (__bf16)dk[(size_t)h * ATT + a];
}

// ---------------------------------------------------------------------------
// Kernel 2: ctx[b,a] = (genes[b,:]*w0) @ w_genes + b_genes[a] + dense_bias[a]
// ---------------------------------------------------------------------------
__global__ void ctx_kernel(const float* __restrict__ genes,
                           const float* __restrict__ w0p,
                           const float* __restrict__ w_genes,
                           const float* __restrict__ b_genes,
                           const float* __restrict__ dense_bias,
                           float* __restrict__ ctx) {
    __shared__ float sg[GENES];
    int b = blockIdx.x, a = threadIdx.x;
    float w0 = w0p[0];
    for (int i = a; i < GENES; i += 256)
        sg[i] = genes[(size_t)b * GENES + i] * w0;
    __syncthreads();
    float acc = b_genes[a] + dense_bias[a];
    for (int g = 0; g < GENES; ++g)
        acc = fmaf(sg[g], w_genes[(size_t)g * ATT + a], acc);
    ctx[(size_t)b * ATT + a] = acc;
}

// ---------------------------------------------------------------------------
// A-operand build: 16x32 bf16 tile per ISA 7.12.2 (16-bit A):
//   lane l: row m = l%16 ; elem e: k = (e/8)*16 + (l/16)*8 + e%8
// -> per lane two contiguous 8-float runs at klocal = (l/16)*8 and 16+(l/16)*8
// ---------------------------------------------------------------------------
__device__ __forceinline__ v16bf load_a_tile(const float* __restrict__ p) {
    const float4* q = (const float4*)p;
    float4 r0 = q[0], r1 = q[1];   // klocal +0..7
    float4 r2 = q[4], r3 = q[5];   // klocal +16..23
    v16bf a;
    a[0]  = (__bf16)r0.x; a[1]  = (__bf16)r0.y; a[2]  = (__bf16)r0.z; a[3]  = (__bf16)r0.w;
    a[4]  = (__bf16)r1.x; a[5]  = (__bf16)r1.y; a[6]  = (__bf16)r1.z; a[7]  = (__bf16)r1.w;
    a[8]  = (__bf16)r2.x; a[9]  = (__bf16)r2.y; a[10] = (__bf16)r2.z; a[11] = (__bf16)r2.w;
    a[12] = (__bf16)r3.x; a[13] = (__bf16)r3.y; a[14] = (__bf16)r3.z; a[15] = (__bf16)r3.w;
    return a;
}

// ---------------------------------------------------------------------------
// Main fused kernel: 1 block per batch element b, 256 threads (8 wave32).
// Wave grid: wn = wave&1 -> N-half (8 tiles of 16 cols), wm = wave>>1 -> M-strip
// of 32 rows (2 tiles). 8 outer iterations cover T=1024 rows.
// ---------------------------------------------------------------------------
__global__ __launch_bounds__(256, 1)
void attn_main_kernel(const float* __restrict__ smiles,
                      const float* __restrict__ ctx,
                      const __bf16* __restrict__ packB,
                      const float* __restrict__ v,
                      float* __restrict__ out,
                      float* __restrict__ alphas_out) {
    __shared__ float lds_xv[TT];
    __shared__ float lds_ctx[ATT];
    __shared__ float lds_v[ATT];
    __shared__ float lds_red[256];

    const int b    = blockIdx.x;
    const int tid  = threadIdx.x;
    const int lane = tid & 31;
    const int wave = tid >> 5;
    const int wn   = wave & 1;   // N half
    const int wm   = wave >> 1;  // M strip
    const int nl   = lane & 15;
    const int hl   = lane >> 4;

    for (int t = tid; t < TT; t += 256) lds_xv[t] = 0.0f;
    lds_ctx[tid] = ctx[(size_t)b * ATT + tid];
    lds_v[tid]   = v[tid];
    __syncthreads();

    const float* srow = smiles + (size_t)b * TT * HID;

#pragma unroll 1
    for (int iter = 0; iter < 8; ++iter) {
        const int trowbase = iter * 128 + wm * 32;
        v8f acc[16];
#pragma unroll
        for (int i = 0; i < 16; ++i) acc[i] = 0.0f;

        const float* a0base = srow + (size_t)(trowbase + nl)      * HID + hl * 8;
        const float* a1base = srow + (size_t)(trowbase + 16 + nl) * HID + hl * 8;

#pragma unroll 4
        for (int ks = 0; ks < 16; ++ks) {
            v16bf a0 = load_a_tile(a0base + ks * 32);
            v16bf a1 = load_a_tile(a1base + ks * 32);
#pragma unroll
            for (int j = 0; j < 8; ++j) {
                const int nt = wn * 8 + j;
                const v16bf bm = *(const v16bf*)(packB + ((size_t)(nt * 16 + ks) * 32 + lane) * 16);
                acc[j]     = __builtin_amdgcn_wmma_f32_16x16x32_bf16(
                                 false, a0, false, bm, (short)0, acc[j],     false, false);
                acc[j + 8] = __builtin_amdgcn_wmma_f32_16x16x32_bf16(
                                 false, a1, false, bm, (short)0, acc[j + 8], false, false);
            }
        }

        // epilogue: xv[t] += sum_a tanh(proj + ctx[a]) * v[a]
        // C/D layout: acc[tile][r] @ lane l -> M = r + 8*(l/16), N = l%16
#pragma unroll
        for (int mt = 0; mt < 2; ++mt) {
#pragma unroll
            for (int r = 0; r < 8; ++r) {
                float s = 0.0f;
#pragma unroll
                for (int j = 0; j < 8; ++j) {
                    const int a = (wn * 8 + j) * 16 + nl;
                    const float x = acc[mt * 8 + j][r];
                    s += tanhf(x + lds_ctx[a]) * lds_v[a];
                }
                // reduce across the 16 lanes of the same half (different a)
                s += __shfl_xor(s, 1);
                s += __shfl_xor(s, 2);
                s += __shfl_xor(s, 4);
                s += __shfl_xor(s, 8);
                if (nl == 0)
                    atomicAdd(&lds_xv[trowbase + mt * 16 + hl * 8 + r], s);
            }
        }
    }
    __syncthreads();

    // softmax over T=1024 in LDS
    float m = -__builtin_inff();
    for (int t = tid; t < TT; t += 256) m = fmaxf(m, lds_xv[t]);
    lds_red[tid] = m;
    __syncthreads();
    for (int s2 = 128; s2 > 0; s2 >>= 1) {
        if (tid < s2) lds_red[tid] = fmaxf(lds_red[tid], lds_red[tid + s2]);
        __syncthreads();
    }
    const float gmax = lds_red[0];
    __syncthreads();

    float psum = 0.0f;
    for (int t = tid; t < TT; t += 256) {
        float e = __expf(lds_xv[t] - gmax);
        lds_xv[t] = e;
        psum += e;
    }
    lds_red[tid] = psum;
    __syncthreads();
    for (int s2 = 128; s2 > 0; s2 >>= 1) {
        if (tid < s2) lds_red[tid] += lds_red[tid + s2];
        __syncthreads();
    }
    const float inv = 1.0f / lds_red[0];
    __syncthreads();

    for (int t = tid; t < TT; t += 256) {
        float al = lds_xv[t] * inv;
        lds_xv[t] = al;
        alphas_out[(size_t)b * TT + t] = al;
    }
    __syncthreads();

    // output[b,h] = sum_t alphas[t] * smiles[b,t,h]
    float o0 = 0.0f, o1 = 0.0f;
#pragma unroll 4
    for (int t = 0; t < TT; ++t) {
        const float al = lds_xv[t];
        o0 = fmaf(al, srow[(size_t)t * HID + tid], o0);
        o1 = fmaf(al, srow[(size_t)t * HID + tid + 256], o1);
    }
    out[(size_t)b * HID + tid]       = o0;
    out[(size_t)b * HID + tid + 256] = o1;
}

// ---------------------------------------------------------------------------
extern "C" void kernel_launch(void* const* d_in, const int* in_sizes, int n_in,
                              void* d_out, int out_size, void* d_ws, size_t ws_size,
                              hipStream_t stream) {
    (void)in_sizes; (void)n_in; (void)out_size; (void)ws_size;
    const float* genes      = (const float*)d_in[0];
    const float* smiles     = (const float*)d_in[1];
    const float* w0         = (const float*)d_in[2];
    const float* w_genes    = (const float*)d_in[3];
    const float* b_genes    = (const float*)d_in[4];
    const float* dk         = (const float*)d_in[5];
    const float* dense_bias = (const float*)d_in[6];
    const float* vv         = (const float*)d_in[7];

    __bf16* packB = (__bf16*)d_ws;                                  // 512*256*2 = 256 KB
    float*  ctx   = (float*)((char*)d_ws + (size_t)HID * ATT * 2);  // 128*256*4 = 128 KB

    float* out_p    = (float*)d_out;            // [128,512]
    float* alphas_p = out_p + (size_t)BB * HID; // [128,1024]

    pack_b_kernel<<<dim3(HID), dim3(ATT), 0, stream>>>(dk, packB);
    ctx_kernel<<<dim3(BB), dim3(ATT), 0, stream>>>(genes, w0, w_genes, b_genes, dense_bias, ctx);
    attn_main_kernel<<<dim3(BB), dim3(256), 0, stream>>>(smiles, ctx, packB, vv, out_p, alphas_p);
}
